// GRNN_3427383902783
// MI455X (gfx1250) — matrix operand order repaired and checked by reading, not compiled, not run on hardware
//
#include <hip/hip_runtime.h>
#include <hip/hip_bf16.h>
#include <math.h>

// ---------------------------------------------------------------------------
// CDNA5 (gfx1250) implementation. wave32. All matrix math via
// v_wmma_f32_16x16x32_bf16 (bf16 inputs, f32 accumulate).
// ---------------------------------------------------------------------------

typedef __attribute__((ext_vector_type(16))) __bf16 v16bf;
typedef __attribute__((ext_vector_type(8)))  __bf16 v8bf;
typedef __attribute__((ext_vector_type(4)))  __bf16 v4bf;
typedef __attribute__((ext_vector_type(8)))  float  v8f;

#define WMMA_BF16(A, B, C) \
  __builtin_amdgcn_wmma_f32_16x16x32_bf16(false, (A), false, (B), (short)0, (C), false, false)

// Zero-pad rows of x (M x K) into xp (M x Kp), Kp % 32 == 0. Coalesced b32.
__global__ void pad_kernel(const float* __restrict__ x, float* __restrict__ xp,
                           int M, int K, int Kp)
{
  int idx = blockIdx.x * blockDim.x + threadIdx.x;
  if (idx >= M * Kp) return;
  int row = idx / Kp, k = idx - row * Kp;
  xp[idx] = (k < K) ? x[(size_t)row * K + k] : 0.f;
}

// ---------------------------------------------------------------------------
// GEMM: C[M,N] = act(A[M,K] @ B[K,N] + bias), act: 0=none, 1=leaky(0.01)
// Preconditions: M % 64 == 0, K % 32 == 0, N % 16 == 0 (driver guarantees).
// Tile BM=64 x BN=64, BK=32; 128 threads (4 waves); wave w owns rows
// [16w,16w+16) x 4 column tiles. Staging branch-free and vectorized.
// ---------------------------------------------------------------------------
#define BM 64
#define BN 64
#define BK 32

__global__ __launch_bounds__(128) void gemm_bf16_kernel(
    const float* __restrict__ A, const float* __restrict__ B,
    const float* __restrict__ bias, float* __restrict__ C,
    int M, int N, int K, int act)
{
  __shared__ __align__(16) __bf16 As[BM * BK];   // row-major [row][k]
  __shared__ __align__(16) __bf16 Bs[BN * BK];   // col-major [col][k]

  const int tid  = threadIdx.x;
  const int lane = tid & 31;
  const int wave = tid >> 5;
  const int m0   = blockIdx.x * BM;
  const int n0   = blockIdx.y * BN;

  const int cq     = (tid & 15) << 2;            // column group base (0..60)
  const int gn     = n0 + cq;
  const bool colok = (gn + 3) < N;
  const int gnc    = colok ? gn : 0;

  v8f acc[4] = {};

  for (int k0 = 0; k0 < K; k0 += BK) {
    #pragma unroll
    for (int g = 0; g < 4; ++g) {
      int gid = tid + g * 128;
      int row = gid >> 3;
      int kq  = (gid & 7) << 2;
      float4 v = *(const float4*)(&A[(size_t)(m0 + row) * K + k0 + kq]);
      v4bf o;
      o[0] = (__bf16)v.x; o[1] = (__bf16)v.y;
      o[2] = (__bf16)v.z; o[3] = (__bf16)v.w;
      *(v4bf*)(&As[row * BK + kq]) = o;
    }
    #pragma unroll
    for (int g = 0; g < 4; ++g) {
      int kk = (tid >> 4) + g * 8;
      float4 v = *(const float4*)(&B[(size_t)(k0 + kk) * N + gnc]);
      Bs[(cq + 0) * BK + kk] = (__bf16)(colok ? v.x : 0.f);
      Bs[(cq + 1) * BK + kk] = (__bf16)(colok ? v.y : 0.f);
      Bs[(cq + 2) * BK + kk] = (__bf16)(colok ? v.z : 0.f);
      Bs[(cq + 3) * BK + kk] = (__bf16)(colok ? v.w : 0.f);
    }
    __syncthreads();

    const int arow = wave * 16 + (lane & 15);
    const int akb  = (lane >> 4) * 8;
    v16bf a;
    {
      const v8bf lo = *(const v8bf*)(&As[arow * BK + akb]);
      const v8bf hi = *(const v8bf*)(&As[arow * BK + akb + 16]);
      #pragma unroll
      for (int i = 0; i < 8; ++i) { a[i] = lo[i]; a[i + 8] = hi[i]; }
    }
    v16bf bf[4];
    const int bk = (lane >> 4) * 8;
    #pragma unroll
    for (int t = 0; t < 4; ++t) {
      const int bcol = t * 16 + (lane & 15);
      const v8bf b0 = *(const v8bf*)(&Bs[bcol * BK + bk]);
      const v8bf b1 = *(const v8bf*)(&Bs[bcol * BK + bk + 16]);
      #pragma unroll
      for (int i = 0; i < 8; ++i) { bf[t][i] = b0[i]; bf[t][i + 8] = b1[i]; }
    }
    #pragma unroll
    for (int t = 0; t < 4; ++t) acc[t] = WMMA_BF16(a, bf[t], acc[t]);
    __syncthreads();
  }

  const int mr = wave * 16;
  #pragma unroll
  for (int t = 0; t < 4; ++t) {
    const int col = n0 + t * 16 + (lane & 15);
    if (col >= N) continue;
    const float bv = bias ? bias[col] : 0.f;
    #pragma unroll
    for (int e = 0; e < 8; ++e) {
      const int row = m0 + mr + e + ((lane >> 4) << 3);
      float v = acc[t][e] + bv;
      if (act == 1) v = (v >= 0.f) ? v : 0.01f * v;
      C[(size_t)row * N + col] = v;
    }
  }
}

// hsrc[i] = h[i,:] . a[0:D] ; hdst[i] = h[i,:] . a[D:2D]
__global__ void rowdot2_kernel(const float* __restrict__ h, const float* __restrict__ a,
                               float* __restrict__ src, float* __restrict__ dst,
                               int N, int D)
{
  int i = blockIdx.x * blockDim.x + threadIdx.x;
  if (i >= N) return;
  float s0 = 0.f, s1 = 0.f;
  const float* row = h + (size_t)i * D;
  for (int k = 0; k < D; k += 4) {
    float4 v  = *(const float4*)(&row[k]);
    float4 a0 = *(const float4*)(&a[k]);
    float4 a1 = *(const float4*)(&a[D + k]);
    s0 += v.x * a0.x + v.y * a0.y + v.z * a0.z + v.w * a0.w;
    s1 += v.x * a1.x + v.y * a1.y + v.z * a1.z + v.w * a1.w;
  }
  src[i] = s0; dst[i] = s1;
}

// Ht[d][n] = bf16(h[n][d]) — write-coalesced (N=4096, D=128)
__global__ void trans_bf16_kernel(const float* __restrict__ h, __bf16* __restrict__ ht,
                                  int N, int D)
{
  int idx = blockIdx.x * blockDim.x + threadIdx.x;
  if (idx >= N * D) return;
  int d = idx / N, n = idx - d * N;
  ht[idx] = (__bf16)h[(size_t)n * D + d];
}

// ---------------------------------------------------------------------------
// Fused masked additive-attention (flash style), one head.
//   e_ij = leaky(hsrc[i]+hdst[j], alpha); masked by adj; row softmax;
//   out = rownorm(leaky(P @ H, alpha)) + bias
// Block: 128 threads (4 waves), 16 rows; stream 32 columns/iter.
// Wave w accumulates output channels [32w, 32w+32) via two bf16 WMMAs/iter.
// adj (64MB int32) stays L2-resident (192MB) across all 16 passes;
// global_prefetch_b8 hides the L2 hit latency for the next tile.
// ---------------------------------------------------------------------------
#define AT_RB 16
#define AT_CB 32

__global__ __launch_bounds__(128) void gat_attn_kernel(
    const __bf16* __restrict__ Ht, const float* __restrict__ hsrc,
    const float* __restrict__ hdst, const int* __restrict__ adj,
    const float* __restrict__ bias, float* __restrict__ out,
    int N, int D, float alpha)
{
  __shared__ __align__(16) float  sc[AT_RB * AT_CB];
  __shared__ __align__(16) __bf16 pb[AT_RB * AT_CB];
  __shared__ float rowm[AT_RB], rowl[AT_RB], corr[AT_RB];
  __shared__ __align__(16) float otile[AT_RB * 128];
  __shared__ float norms[AT_RB];

  const int tid  = threadIdx.x;
  const int lane = tid & 31;
  const int wave = tid >> 5;
  const int r0   = blockIdx.x * AT_RB;

  if (tid < AT_RB) { rowm[tid] = -3.0e38f; rowl[tid] = 0.f; }
  v8f acc0 = {}, acc1 = {};

  const int sr   = tid >> 3;
  const int sc4  = (tid & 7) << 2;
  const float hs = hsrc[r0 + sr];
  const int* adjrow = adj + (size_t)(r0 + sr) * N;
  const __bf16* htp0 = Ht + (size_t)(wave * 32 + (lane & 15)) * N + ((lane >> 4) << 3);
  const __bf16* htp1 = htp0 + (size_t)16 * N;
  __syncthreads();

  for (int jb = 0; jb < N; jb += AT_CB) {
    // Prefetch next tile's streams (L2 -> L0) while we work on this one.
    if (jb + AT_CB < N) {
      __builtin_prefetch(&adjrow[jb + AT_CB + sc4], 0, 1);
      __builtin_prefetch(htp0 + jb + AT_CB, 0, 1);
      __builtin_prefetch(htp1 + jb + AT_CB, 0, 1);
    }
    // Phase A: masked leaky scores, vectorized (int4 adj, float4 hdst/sc).
    {
      const int gj = jb + sc4;
      const int4   m4 = *(const int4*)(&adjrow[gj]);
      const float4 hd = *(const float4*)(&hdst[gj]);
      float4 s;
      float t;
      t = hs + hd.x; t = (t >= 0.f) ? t : alpha * t; s.x = (m4.x > 0) ? t : -9.0e15f;
      t = hs + hd.y; t = (t >= 0.f) ? t : alpha * t; s.y = (m4.y > 0) ? t : -9.0e15f;
      t = hs + hd.z; t = (t >= 0.f) ? t : alpha * t; s.z = (m4.z > 0) ? t : -9.0e15f;
      t = hs + hd.w; t = (t >= 0.f) ? t : alpha * t; s.w = (m4.w > 0) ? t : -9.0e15f;
      *(float4*)(&sc[sr * AT_CB + sc4]) = s;
    }
    __syncthreads();

    // Phase B: online-softmax (wave 0; lane pair per row + shfl reduction).
    if (wave == 0) {
      const int r = lane >> 1;
      const int hh = (lane & 1) << 4;
      float tmax = -3.0e38f;
      #pragma unroll
      for (int c = 0; c < 16; ++c) tmax = fmaxf(tmax, sc[r * AT_CB + hh + c]);
      tmax = fmaxf(tmax, __shfl_xor(tmax, 1, 32));
      const float m  = rowm[r];
      const float nm = fmaxf(m, tmax);
      float sum = 0.f;
      #pragma unroll
      for (int c = 0; c < 16; ++c) {
        float p = __expf(sc[r * AT_CB + hh + c] - nm);
        pb[r * AT_CB + hh + c] = (__bf16)p;
        sum += p;
      }
      sum += __shfl_xor(sum, 1, 32);
      if ((lane & 1) == 0) {
        const float cr = __expf(m - nm);
        corr[r] = cr;
        rowm[r] = nm;
        rowl[r] = rowl[r] * cr + sum;
      }
    }
    __syncthreads();

    // Phase C: rescale accumulators, then acc += P @ H_block.
    const int arow = lane & 15;
    const int akb  = (lane >> 4) * 8;
    v16bf a;
    {
      const v8bf lo = *(const v8bf*)(&pb[arow * AT_CB + akb]);
      const v8bf hi = *(const v8bf*)(&pb[arow * AT_CB + akb + 16]);
      #pragma unroll
      for (int i = 0; i < 8; ++i) { a[i] = lo[i]; a[i + 8] = hi[i]; }
    }
    #pragma unroll
    for (int e = 0; e < 8; ++e) {
      float cr = corr[e + ((lane >> 4) << 3)];
      acc0[e] *= cr; acc1[e] *= cr;
    }
    {
      const v8bf b0 = *(const v8bf*)(htp0 + jb);
      const v8bf b1 = *(const v8bf*)(htp0 + jb + 16);
      const v8bf c0 = *(const v8bf*)(htp1 + jb);
      const v8bf c1 = *(const v8bf*)(htp1 + jb + 16);
      v16bf b, b2;
      #pragma unroll
      for (int i = 0; i < 8; ++i) {
        b[i]  = b0[i]; b[i + 8]  = b1[i];
        b2[i] = c0[i]; b2[i + 8] = c1[i];
      }
      acc0 = WMMA_BF16(a, b,  acc0);
      acc1 = WMMA_BF16(a, b2, acc1);
    }
    __syncthreads();
  }

  // Epilogue: softmax normalize, leaky, stash to LDS for row L2-norm.
  #pragma unroll
  for (int e = 0; e < 8; ++e) {
    const int r = e + ((lane >> 4) << 3);
    const float inv = 1.f / rowl[r];
    float v0 = acc0[e] * inv; v0 = (v0 >= 0.f) ? v0 : alpha * v0;
    float v1 = acc1[e] * inv; v1 = (v1 >= 0.f) ? v1 : alpha * v1;
    otile[r * 128 + wave * 32 + (lane & 15)]      = v0;
    otile[r * 128 + wave * 32 + 16 + (lane & 15)] = v1;
  }
  __syncthreads();
  if (tid < AT_RB) {
    float s = 0.f;
    for (int c = 0; c < 128; c += 4) {
      float4 v = *(const float4*)(&otile[tid * 128 + c]);
      s += v.x * v.x + v.y * v.y + v.z * v.z + v.w * v.w;
    }
    norms[tid] = fmaxf(sqrtf(s), 1e-12f);
  }
  __syncthreads();
  #pragma unroll
  for (int q = 0; q < 4; ++q) {
    int idx = (tid + q * 128) << 2;
    int r = idx >> 7;
    int c = idx & 127;
    float4 v  = *(const float4*)(&otile[idx]);
    float4 bv = *(const float4*)(&bias[c]);
    const float inv = 1.f / norms[r];
    float4 o;
    o.x = v.x * inv + bv.x; o.y = v.y * inv + bv.y;
    o.z = v.z * inv + bv.z; o.w = v.w * inv + bv.w;
    *(float4*)(&out[(size_t)(r0 + r) * D + c]) = o;
  }
}

__global__ void combine2_kernel(const float* __restrict__ h0, const float* __restrict__ h1,
                                float* __restrict__ o, int n, int do_elu)
{
  int i = blockIdx.x * blockDim.x + threadIdx.x;
  if (i >= n) return;
  float v = 0.5f * (h0[i] + h1[i]);
  if (do_elu) v = (v > 0.f) ? v : (__expf(v) - 1.f);
  o[i] = v;
}

__global__ void gather_kernel(const float* __restrict__ tf, const float* __restrict__ tg,
                              const int* __restrict__ samp, float* __restrict__ step, int E)
{
  int idx = blockIdx.x * blockDim.x + threadIdx.x;
  if (idx >= E * 32) return;
  int e = idx >> 5, c = idx & 31;
  int node = samp[e * 2 + (c >= 16 ? 1 : 0)];
  step[idx] = (c < 16) ? tf[(size_t)node * 16 + c] : tg[(size_t)node * 16 + (c - 16)];
}

__global__ void gru_gate_kernel(const float* __restrict__ gi, const float* __restrict__ gh,
                                const float* __restrict__ hprev, float* __restrict__ hnew,
                                int B, int H)
{
  int idx = blockIdx.x * blockDim.x + threadIdx.x;
  if (idx >= B * H) return;
  int b = idx / H, u = idx - b * H;
  const float* gib = gi + (size_t)b * 3 * H;
  const float* ghb = gh + (size_t)b * 3 * H;
  float r = 1.f / (1.f + __expf(-(gib[u] + ghb[u])));
  float z = 1.f / (1.f + __expf(-(gib[H + u] + ghb[H + u])));
  float n = tanhf(gib[2 * H + u] + r * ghb[2 * H + u]);
  hnew[idx] = (1.f - z) * n + z * hprev[idx];
}

// softmax over T per (b,c), weighted sum of ys (values cached in registers)
__global__ void tpool_kernel(const float* __restrict__ s2, const float* __restrict__ ys,
                             float* __restrict__ o, int B, int C)
{
  const int T = 4;
  int idx = blockIdx.x * blockDim.x + threadIdx.x;
  if (idx >= B * C) return;
  int b = idx / C, c = idx - b * C;
  float sv[T], yv[T];
  #pragma unroll
  for (int t = 0; t < T; ++t) {
    sv[t] = s2[((size_t)t * B + b) * C + c];
    yv[t] = ys[((size_t)t * B + b) * C + c];
  }
  float m = fmaxf(fmaxf(sv[0], sv[1]), fmaxf(sv[2], sv[3]));
  float sum = 0.f, acc = 0.f;
  #pragma unroll
  for (int t = 0; t < T; ++t) {
    float p = __expf(sv[t] - m);
    sum += p;
    acc += p * yv[t];
  }
  o[idx] = acc / sum;
}

__global__ void final_kernel(const float* __restrict__ att, const float* __restrict__ fow,
                             const float* __restrict__ fob, float* __restrict__ o,
                             int B, int C)
{
  int b = blockIdx.x * blockDim.x + threadIdx.x;
  if (b >= B) return;
  float s = 0.f;
  for (int c = 0; c < C; c += 4) {
    float4 v = *(const float4*)(&att[(size_t)b * C + c]);
    float4 w = *(const float4*)(&fow[c]);
    s += v.x * w.x + v.y * w.y + v.z * w.z + v.w * w.w;
  }
  o[b] = s + fob[0];
}

__global__ void zero_kernel(float* __restrict__ p, int n)
{
  int i = blockIdx.x * blockDim.x + threadIdx.x;
  if (i < n) p[i] = 0.f;
}

// ---------------------------------------------------------------------------
// Orchestration
// ---------------------------------------------------------------------------
extern "C" void kernel_launch(void* const* d_in, const int* in_sizes, int n_in,
                              void* d_out, int out_size, void* d_ws, size_t ws_size,
                              hipStream_t stream)
{
  (void)in_sizes; (void)n_in; (void)out_size; (void)ws_size;

  const int N = 4096, E = 4096, IN_DIM = 256, H1d = 128, H2d = 128, H3d = 64;
  const int OUTD = 16, GH = 128, T = 4;
  const int TD[4]  = {933, 303, 683, 798};
  const int TDP[4] = {960, 320, 704, 800};     // K padded to multiples of 32
  const float ALPHA = 0.2f;

  const int*   adj  = (const int*)d_in[4];
  const int*   samp = (const int*)d_in[5];
  const float* W1   = (const float*)d_in[14];
  const float* a1   = (const float*)d_in[15];
  const float* b1   = (const float*)d_in[16];
  const float* W2   = (const float*)d_in[17];
  const float* a2   = (const float*)d_in[18];
  const float* b2   = (const float*)d_in[19];
  const float* tfw1 = (const float*)d_in[20]; const float* tfb1 = (const float*)d_in[21];
  const float* tfw2 = (const float*)d_in[22]; const float* tfb2 = (const float*)d_in[23];
  const float* tgw1 = (const float*)d_in[24]; const float* tgb1 = (const float*)d_in[25];
  const float* tgw2 = (const float*)d_in[26]; const float* tgb2 = (const float*)d_in[27];
  const float* wih0 = (const float*)d_in[28]; const float* whh0 = (const float*)d_in[29];
  const float* bih0 = (const float*)d_in[30]; const float* bhh0 = (const float*)d_in[31];
  const float* wih1 = (const float*)d_in[32]; const float* whh1 = (const float*)d_in[33];
  const float* bih1 = (const float*)d_in[34]; const float* bhh1 = (const float*)d_in[35];
  const float* attw = (const float*)d_in[36]; const float* attb = (const float*)d_in[37];
  const float* fow  = (const float*)d_in[38]; const float* fob  = (const float*)d_in[39];

  char* ws = (char*)d_ws;
  const size_t MB = 1u << 20;
  float*  e_buf  = (float*)(ws + 0 * MB);
  float*  hb0    = (float*)(ws + 4 * MB);
  float*  hb1    = (float*)(ws + 6 * MB);
  __bf16* ht0    = (__bf16*)(ws + 8 * MB);
  __bf16* ht1    = (__bf16*)(ws + 9 * MB);
  float*  sd     = (float*)(ws + 10 * MB);
  float*  ao0    = (float*)(ws + 11 * MB);
  float*  ao1    = (float*)(ws + 13 * MB);
  float*  x1b    = (float*)(ws + 15 * MB);
  float*  emb    = (float*)(ws + 17 * MB);
  float*  tf1    = (float*)(ws + 19 * MB);
  float*  tfv    = (float*)(ws + 20 * MB);
  float*  tg1    = (float*)(ws + 21 * MB);
  float*  tgv    = (float*)(ws + 22 * MB);
  float*  steps  = (float*)(ws + 23 * MB);
  float*  gi     = (float*)(ws + 25 * MB);
  float*  gh     = (float*)(ws + 31 * MB);
  float*  ys0    = (float*)(ws + 37 * MB);
  float*  ys1    = (float*)(ws + 45 * MB);
  float*  s2     = (float*)(ws + 53 * MB);
  float*  attout = (float*)(ws + 61 * MB);
  float*  hzero  = (float*)(ws + 63 * MB);
  float*  xpad   = (float*)(ws + 65 * MB);     // 4096x960 max (15 MB)
  float*  bpad   = (float*)(ws + 81 * MB);     // 960x256 max (<1 MB)

  auto gemm = [&](const float* A, const float* B, const float* bias, float* C,
                  int M, int Nn, int K, int act) {
    dim3 g(M / BM, (Nn + BN - 1) / BN);
    gemm_bf16_kernel<<<g, 128, 0, stream>>>(A, B, bias, C, M, Nn, K, act);
  };

  zero_kernel<<<(N * GH + 255) / 256, 256, 0, stream>>>(hzero, N * GH);

  for (int st = 0; st < 4; ++st) {
    const float* x  = (const float*)d_in[st];
    const float* nw = (const float*)d_in[6 + 2 * st];
    const float* nb = (const float*)d_in[7 + 2 * st];
    const int K = TD[st], Kp = TDP[st];

    // Pad A columns (ragged K -> K%32==0); pad B rows (zero tail + d2d copy).
    pad_kernel<<<(N * Kp + 255) / 256, 256, 0, stream>>>(x, xpad, N, K, Kp);
    zero_kernel<<<((Kp - K) * IN_DIM + 255) / 256, 256, 0, stream>>>(
        bpad + (size_t)K * IN_DIM, (Kp - K) * IN_DIM);
    hipMemcpyAsync(bpad, nw, (size_t)K * IN_DIM * sizeof(float),
                   hipMemcpyDeviceToDevice, stream);
    gemm(xpad, bpad, nb, e_buf, N, IN_DIM, Kp, 0);

    for (int hd = 0; hd < 2; ++hd) {
      const float* Wk = W1 + (size_t)hd * IN_DIM * H1d;
      const float* ak = a1 + (size_t)hd * 2 * H1d;
      const float* bk = b1 + (size_t)hd * H1d;
      float*  h   = hd ? hb1 : hb0;
      __bf16* htk = hd ? ht1 : ht0;
      float*  ao  = hd ? ao1 : ao0;
      float*  hsk = sd + (size_t)hd * 2 * N;
      float*  hdk = hsk + N;
      gemm(e_buf, Wk, nullptr, h, N, H1d, IN_DIM, 0);
      rowdot2_kernel<<<(N + 255) / 256, 256, 0, stream>>>(h, ak, hsk, hdk, N, H1d);
      trans_bf16_kernel<<<(N * H1d + 255) / 256, 256, 0, stream>>>(h, htk, N, H1d);
      gat_attn_kernel<<<dim3(N / AT_RB), 128, 0, stream>>>(htk, hsk, hdk, adj, bk, ao,
                                                           N, H1d, ALPHA);
    }
    combine2_kernel<<<(N * H1d + 255) / 256, 256, 0, stream>>>(ao0, ao1, x1b, N * H1d, 1);

    for (int hd = 0; hd < 2; ++hd) {
      const float* Wk = W2 + (size_t)hd * H1d * H2d;
      const float* ak = a2 + (size_t)hd * 2 * H2d;
      const float* bk = b2 + (size_t)hd * H2d;
      float*  h   = hd ? hb1 : hb0;
      __bf16* htk = hd ? ht1 : ht0;
      float*  ao  = hd ? ao1 : ao0;
      float*  hsk = sd + (size_t)hd * 2 * N;
      float*  hdk = hsk + N;
      gemm(x1b, Wk, nullptr, h, N, H2d, H1d, 0);
      rowdot2_kernel<<<(N + 255) / 256, 256, 0, stream>>>(h, ak, hsk, hdk, N, H2d);
      trans_bf16_kernel<<<(N * H2d + 255) / 256, 256, 0, stream>>>(h, htk, N, H2d);
      gat_attn_kernel<<<dim3(N / AT_RB), 128, 0, stream>>>(htk, hsk, hdk, adj, bk, ao,
                                                           N, H2d, ALPHA);
    }
    combine2_kernel<<<(N * H2d + 255) / 256, 256, 0, stream>>>(ao0, ao1, emb, N * H2d, 0);

    gemm(emb, tfw1, tfb1, tf1, N, H3d, H2d, 1);
    gemm(tf1, tfw2, tfb2, tfv, N, OUTD, H3d, 1);
    gemm(emb, tgw1, tgb1, tg1, N, H3d, H2d, 1);
    gemm(tg1, tgw2, tgb2, tgv, N, OUTD, H3d, 1);

    gather_kernel<<<(E * 32 + 255) / 256, 256, 0, stream>>>(
        tfv, tgv, samp, steps + (size_t)st * E * 32, E);
  }

  for (int t = 0; t < T; ++t) {
    const float* xin = steps + (size_t)t * N * 32;
    const float* hp  = (t == 0) ? hzero : (ys0 + (size_t)(t - 1) * N * GH);
    gemm(xin, wih0, bih0, gi, N, 3 * GH, 32, 0);
    gemm(hp, whh0, bhh0, gh, N, 3 * GH, GH, 0);
    gru_gate_kernel<<<(N * GH + 255) / 256, 256, 0, stream>>>(
        gi, gh, hp, ys0 + (size_t)t * N * GH, N, GH);
  }
  for (int t = 0; t < T; ++t) {
    const float* xin = ys0 + (size_t)t * N * GH;
    const float* hp  = (t == 0) ? hzero : (ys1 + (size_t)(t - 1) * N * GH);
    gemm(xin, wih1, bih1, gi, N, 3 * GH, GH, 0);
    gemm(hp, whh1, bhh1, gh, N, 3 * GH, GH, 0);
    gru_gate_kernel<<<(N * GH + 255) / 256, 256, 0, stream>>>(
        gi, gh, hp, ys1 + (size_t)t * N * GH, N, GH);
  }

  gemm(ys1, attw, attb, s2, T * N, GH, GH, 0);
  tpool_kernel<<<(N * GH + 255) / 256, 256, 0, stream>>>(s2, ys1, attout, N, GH);
  final_kernel<<<(N + 255) / 256, 256, 0, stream>>>(attout, fow, fob, (float*)d_out, N, GH);
}
